// GatedBlock_12927851561145
// MI455X (gfx1250) — matrix operand order repaired and checked by reading, not compiled, not run on hardware
//
#include <hip/hip_runtime.h>
#include <hip/hip_bf16.h>
#include <math.h>

// ---------------------------------------------------------------------------
// GatedBlock for MI455X (gfx1250):
//   1) packw : W fp32 -> bf16 [co][tap][ci_pad=96]
//   2) packx : x fp32 -> bf16 halo-padded channels-last Xp[b][36][36][36][96]
//   3) convgate : implicit GEMM on v_wmma_f32_16x16x32_bf16; per-kd input
//      slab is ONE contiguous 55 KB block of Xp, DMA'd to LDS with
//      tensor_load_to_lds (TDM, TENSORcnt) and double-buffered so the DMA of
//      slab kd+1 overlaps the K-loop of slab kd. Gate fused in the epilogue.
//   4) lowpass : 5^3 Gaussian, stride 2.
// ---------------------------------------------------------------------------

typedef __attribute__((ext_vector_type(16))) __bf16 v16bf;
typedef __attribute__((ext_vector_type(8)))  float  v8f;

struct pack32 { uint4 a, b; };   // 32 bytes == v16bf

#define CIN   72
#define CIP   96                  // ci padded to 3 K-chunks of 32
#define COUT  176
#define NFLD  144
#define DIM   32
#define PDIM  36                  // padded spatial extent (halo 2 each side)
#define TAPS  125
#define SLAB_BYTES (8 * PDIM * CIP * 2)           // 55296 B per kd slab
#define SLAB_USH   (8 * PDIM * CIP)               // 27648 ushorts
#define WB_ELEMS   ((size_t)COUT * TAPS * CIP)    // 2,112,000 bf16
#define XP_ELEMS   ((size_t)4 * PDIM * PDIM * PDIM * CIP)  // 17,915,904 bf16

#if defined(__gfx1250__) && __has_builtin(__builtin_amdgcn_tensor_load_to_lds) && \
    __has_builtin(__builtin_amdgcn_s_wait_tensorcnt)
#define USE_TDM 1
#else
#define USE_TDM 0
#endif

// ---------------------------------------------------------------------------
// Kernel 1: W fp32 [co][ci][kd][kh][kw] -> bf16 [co][tap][ci_pad], zero-pad ci
// ---------------------------------------------------------------------------
__global__ __launch_bounds__(256) void packw_kernel(
    const float* __restrict__ W, unsigned short* __restrict__ Wb)
{
    size_t idx = (size_t)blockIdx.x * 256 + threadIdx.x;
    if (idx >= WB_ELEMS) return;
    int ci  = (int)(idx % CIP);
    int tap = (int)((idx / CIP) % TAPS);
    int co  = (int)(idx / ((size_t)CIP * TAPS));
    float v = (ci < CIN) ? W[((size_t)co * CIN + ci) * TAPS + tap] : 0.0f;
    Wb[idx] = __builtin_bit_cast(unsigned short, (__bf16)v);
}

// ---------------------------------------------------------------------------
// Kernel 2: x fp32 [b][ci][z][y][w] -> bf16 Xp[b][zp][yp][wp][ci_pad]
// (halo rows/cols and ci>=72 are zero, so conv staging needs no bounds checks
//  and each per-kd slab is one contiguous block)
// ---------------------------------------------------------------------------
__global__ __launch_bounds__(256) void packx_kernel(
    const float* __restrict__ x, unsigned int* __restrict__ Xp2)
{
    const size_t total = XP_ELEMS / 2;            // uint pairs
    size_t idx = (size_t)blockIdx.x * 256 + threadIdx.x;
    if (idx >= total) return;
    int cip2 = (int)(idx % (CIP / 2));
    size_t t = idx / (CIP / 2);
    int wp = (int)(t % PDIM); t /= PDIM;
    int yp = (int)(t % PDIM); t /= PDIM;
    int zp = (int)(t % PDIM);
    int b  = (int)(t / PDIM);
    int ci = cip2 * 2, z = zp - 2, y = yp - 2, w = wp - 2;
    unsigned int pk = 0u;
    if (z >= 0 && z < DIM && y >= 0 && y < DIM && w >= 0 && w < DIM && ci < CIN) {
        const float* xp = x + (((size_t)(b * CIN + ci) * DIM + z) * DIM + y) * DIM + w;
        __bf16 lo = (__bf16)xp[0];
        __bf16 hi = (ci + 1 < CIN) ? (__bf16)xp[(size_t)DIM * DIM * DIM] : (__bf16)0.f;
        pk = (unsigned int)__builtin_bit_cast(unsigned short, lo) |
             ((unsigned int)__builtin_bit_cast(unsigned short, hi) << 16);
    }
    Xp2[idx] = pk;
}

// ---------------------------------------------------------------------------
// TDM helper: DMA one contiguous slab (6912 x 8B) from global to LDS.
// D# packing per CDNA5 ISA:
//   g0: count=1 | lds_addr[63:32] | global_addr[120:64] | type=2[127:126]
//   g1: data_size=3(8B)[17:16], tensor_dim0[79:48], tensor_dim1=1[111:80],
//       tile_dim0[127:112], tile_dim1/2=0, tensor_dim0_stride[207:160]
// This toolchain's builtin is the 6-arg variant:
//   (u32x4 g0, i32x8 g1, i32x4 g2, i32x4 g3, i32x8 extra, i32 cpol)
// ---------------------------------------------------------------------------
#if USE_TDM
__device__ __forceinline__ void tdm_load_slab(const void* gsrc, unsigned lds_off)
{
    typedef unsigned int u32x4 __attribute__((ext_vector_type(4)));
    typedef int i32x8 __attribute__((ext_vector_type(8)));
    typedef int i32x4 __attribute__((ext_vector_type(4)));
    unsigned long long ga = (unsigned long long)(uintptr_t)gsrc;
    const unsigned n8 = SLAB_BYTES / 8;           // 6912 elements of 8 B
    u32x4 g0;
    g0.x = 1u;                                    // count=1 (valid), user mode
    g0.y = lds_off;                               // LDS byte address
    g0.z = (unsigned)(ga & 0xffffffffu);          // global_addr[31:0]
    g0.w = (unsigned)((ga >> 32) & 0x01ffffffu)   // global_addr[56:32]
         | (2u << 30);                            // type = 2 ("image")
    i32x8 g1;
    g1[0] = (int)(3u << 16);                      // data_size = 8 B
    g1[1] = (int)((n8 & 0xffffu) << 16);          // tensor_dim0[15:0]
    g1[2] = (int)(((n8 >> 16) & 0xffffu)          // tensor_dim0[31:16]
                  | (1u << 16));                  // tensor_dim1 = 1
    g1[3] = (int)((n8 & 0xffffu) << 16);          // tile_dim0 = n8
    g1[4] = 0;                                    // tile_dim1/2 unused
    g1[5] = (int)n8;                              // tensor_dim0_stride
    g1[6] = 0;
    g1[7] = 0;
    i32x4 zg4 = {0, 0, 0, 0};
    i32x8 zg8 = {0, 0, 0, 0, 0, 0, 0, 0};
    __builtin_amdgcn_tensor_load_to_lds(g0, g1, zg4, zg4, zg8, 0);
}
#endif

// ---------------------------------------------------------------------------
// Kernel 3: fused conv + gate.
// block = 352 threads = 11 waves; wave w -> co tile [16w, 16w+16).
// block tile: fixed (b, z), y in [y0,y0+4), w in [0,32) -> 8 WMMA N-tiles.
// Dynamic LDS: two 55296 B slabs (double buffer), bf16 [yrow8][wpad36][ci96].
// ---------------------------------------------------------------------------
__global__ __launch_bounds__(352) void convgate_kernel(
    const unsigned short* __restrict__ Xp,
    const unsigned short* __restrict__ Wb,
    const float* __restrict__ scalar_bias,
    const float* __restrict__ gate_bias,
    float* __restrict__ zbuf)
{
    extern __shared__ __align__(16) unsigned char smem[];
    unsigned short* xs = (unsigned short*)smem;
    float* gsh = (float*)smem;                    // [32 gates][4 y][32 w]

    const int tid  = threadIdx.x;
    const int wave = tid >> 5;
    const int lane = tid & 31;
    const int ln   = lane & 15;    // N / M index within tile
    const int hs   = lane >> 4;    // lane-half select

    int bid = blockIdx.x;          // 0..1023
    const int ytile = bid & 7;  bid >>= 3;
    const int zz    = bid & 31; bid >>= 5;
    const int b     = bid;         // 0..3
    const int y0    = ytile * 4;
    const int coBase = wave * 16;

    // per-kd slab source inside Xp: rows [y0 .. y0+7] at zp = zz+kd, all
    // contiguous: [b][zp][yp][wp][ci]
    auto slab_src = [&](int kd) -> const unsigned short* {
        return Xp + (((size_t)(b * PDIM + (zz + kd)) * PDIM + y0) * PDIM) * CIP;
    };

    v8f acc[8];
#pragma unroll
    for (int i = 0; i < 8; ++i)
        acc[i] = (v8f){0.f, 0.f, 0.f, 0.f, 0.f, 0.f, 0.f, 0.f};

#if USE_TDM
    const unsigned lds0 = (unsigned)(uintptr_t)smem;   // LDS byte offset
    if (wave == 0) {
        tdm_load_slab(slab_src(0), lds0);
        __builtin_amdgcn_s_wait_tensorcnt(0);
    }
    __syncthreads();
    int buf = 0;
#endif

#pragma unroll 1
    for (int kd = 0; kd < 5; ++kd) {
#if USE_TDM
        // kick off DMA of the next slab into the other buffer; it overlaps
        // the whole K-loop below (TENSORcnt waited at the end of this iter).
        if (kd + 1 < 5 && wave == 0)
            tdm_load_slab(slab_src(kd + 1), lds0 + (unsigned)((buf ^ 1) * SLAB_BYTES));
        const int sb = buf * SLAB_USH;
#else
        // synchronous fallback: cooperative contiguous uint4 copy
        __syncthreads();
        {
            const uint4* g4 = (const uint4*)slab_src(kd);
            uint4* l4 = (uint4*)smem;
#pragma unroll 1
            for (int j = tid; j < SLAB_BYTES / 16; j += 352) l4[j] = g4[j];
        }
        __syncthreads();
        const int sb = 0;
#endif

        // ---------------- K loop: 3 ci-chunks x 25 (kh,kw) taps ------------
#pragma unroll 1
        for (int cib = 0; cib < 3; ++cib) {
            const int ci0 = cib * 32;
#pragma unroll 1
            for (int kh = 0; kh < 5; ++kh) {
#pragma unroll
                for (int kw = 0; kw < 5; ++kw) {
                    const int tap = (kd * 5 + kh) * 5 + kw;
                    // A fragment (16co x 32ci), ISA 16-bit A layout:
                    // e0..7 -> K=ci0+hs*8+0..7 ; e8..15 -> K=ci0+16+hs*8+0..7
                    const unsigned short* wp =
                        Wb + ((size_t)(coBase + ln) * TAPS + tap) * CIP + ci0 + hs * 8;
                    __builtin_prefetch(wp + CIP, 0, 3);   // next-tap weights
                    pack32 ap;
                    ap.a = *(const uint4*)(wp);
                    ap.b = *(const uint4*)(wp + 16);
                    v16bf af = __builtin_bit_cast(v16bf, ap);
#pragma unroll
                    for (int nt = 0; nt < 8; ++nt) {
                        const int ny   = nt >> 1;
                        const int wq   = (nt & 1) * 16 + ln + kw; // padded w
                        const int yrow = ny + kh;
                        // B fragment: e0..7 -> K=ci0+hs*8+0..7 ; e8..15 -> K=ci0+16+hs*8+0..7
                        const uint4* bp = (const uint4*)(
                            xs + (sb + (yrow * PDIM + wq) * CIP + ci0 + hs * 8));
                        pack32 bpk; bpk.a = bp[0]; bpk.b = bp[2];
                        v16bf bfm = __builtin_bit_cast(v16bf, bpk);
                        acc[nt] = __builtin_amdgcn_wmma_f32_16x16x32_bf16(
                            false, af, false, bfm, (short)0, acc[nt],
                            false, false);
                    }
                }
            }
        }

#if USE_TDM
        if (wave == 0) __builtin_amdgcn_s_wait_tensorcnt(0);
        __syncthreads();           // next iter reads buf^1 / reissues into buf
        buf ^= 1;
#endif
    }

    // ---------------- epilogue: gate fusion --------------------------------
    __syncthreads();                       // all LDS slab reads complete
    if (wave >= 9) {                       // gate channels 144..175
        const int gbase = (wave - 9) * 16;
#pragma unroll
        for (int nt = 0; nt < 8; ++nt) {
            const int ny = nt >> 1;
            const int w  = (nt & 1) * 16 + ln;
#pragma unroll
            for (int r = 0; r < 8; ++r) {
                const int gc = gbase + r + hs * 8;           // 0..31
                float t = acc[nt][r] + gate_bias[gc];
                gsh[(gc * 4 + ny) * 32 + w] = 1.0f / (1.0f + __expf(-t));
            }
        }
    }
    __syncthreads();
    if (wave < 9) {                        // field channels 0..143
#pragma unroll
        for (int nt = 0; nt < 8; ++nt) {
            const int ny = nt >> 1;
            const int w  = (nt & 1) * 16 + ln;
            const int y  = y0 + ny;
#pragma unroll
            for (int r = 0; r < 8; ++r) {
                const int co = coBase + r + hs * 8;
                float v = acc[nt][r];
                if (co < 16) {                                  // scalars: ReLU
                    v += scalar_bias[co];
                    v = v > 0.f ? v : 0.f;
                } else {                                        // vectors: gate
                    int mul = (co < 64) ? ((co - 16) / 3)
                                        : (16 + (co - 64) / 5);
                    v *= gsh[(mul * 4 + ny) * 32 + w];
                }
                zbuf[((((size_t)b * NFLD + co) * DIM + zz) * DIM + y) * DIM + w] = v;
            }
        }
    }
}

// ---------------------------------------------------------------------------
// Kernel 4: 5x5x5 separable Gaussian (sigma = 0.5*sqrt(3)), stride 2, pad 2.
// ---------------------------------------------------------------------------
__global__ __launch_bounds__(256) void lowpass_kernel(
    const float* __restrict__ z, float* __restrict__ out)
{
    const float g1[5] = {0.0320820f, 0.2370554f, 0.4617229f,
                         0.2370554f, 0.0320820f};
    const size_t total = (size_t)4 * NFLD * 16 * 16 * 16;
    size_t idx = (size_t)blockIdx.x * 256 + threadIdx.x;
    if (idx >= total) return;
    int ow = (int)(idx & 15);
    int oh = (int)((idx >> 4) & 15);
    int od = (int)((idx >> 8) & 15);
    int bc = (int)(idx >> 12);                 // b*144 + c
    const float* zp = z + (size_t)bc * (DIM * DIM * DIM);
    float acc = 0.f;
#pragma unroll
    for (int kd = 0; kd < 5; ++kd) {
        int zi = od * 2 + kd - 2;
        if (zi < 0 || zi >= DIM) continue;
#pragma unroll
        for (int kh = 0; kh < 5; ++kh) {
            int yi = oh * 2 + kh - 2;
            if (yi < 0 || yi >= DIM) continue;
            float gdh = g1[kd] * g1[kh];
#pragma unroll
            for (int kw = 0; kw < 5; ++kw) {
                int wi = ow * 2 + kw - 2;
                if (wi < 0 || wi >= DIM) continue;
                acc += gdh * g1[kw] * zp[(zi * DIM + yi) * DIM + wi];
            }
        }
    }
    out[idx] = acc;
}

// ---------------------------------------------------------------------------
// Launcher. Scratch: [Wb bf16 4.22MB][Xp bf16 35.8MB][zbuf f32 75.5MB]
// ---------------------------------------------------------------------------
extern "C" void kernel_launch(void* const* d_in, const int* in_sizes, int n_in,
                              void* d_out, int out_size, void* d_ws, size_t ws_size,
                              hipStream_t stream) {
    (void)in_sizes; (void)n_in; (void)out_size; (void)ws_size;
    const float* x  = (const float*)d_in[0];
    const float* W  = (const float*)d_in[1];
    const float* sb = (const float*)d_in[2];
    const float* gb = (const float*)d_in[3];

    unsigned short* Wb = (unsigned short*)d_ws;
    unsigned short* Xp = (unsigned short*)((char*)d_ws + WB_ELEMS * sizeof(unsigned short));
    float* zbuf = (float*)((char*)Xp + XP_ELEMS * sizeof(unsigned short));
    float* out  = (float*)d_out;

    {   // 1) weight prepack to bf16, ci padded to 96
        int nblk = (int)((WB_ELEMS + 255) / 256);
        packw_kernel<<<nblk, 256, 0, stream>>>(W, Wb);
    }
    {   // 2) input prepack: halo-padded channels-last bf16
        size_t pairs = XP_ELEMS / 2;
        int nblk = (int)((pairs + 255) / 256);
        packx_kernel<<<nblk, 256, 0, stream>>>(x, (unsigned int*)Xp);
    }
    {   // 3) fused conv + gate (double-buffered TDM staging)
        convgate_kernel<<<4 * 32 * 8, 352, 2 * SLAB_BYTES, stream>>>(
            Xp, Wb, sb, gb, zbuf);
    }
    {   // 4) Gaussian low-pass, stride 2
        size_t total = (size_t)4 * NFLD * 16 * 16 * 16;
        int nblk = (int)((total + 255) / 256);
        lowpass_kernel<<<nblk, 256, 0, stream>>>(zbuf, out);
    }
}
